// attention_76089640615954
// MI455X (gfx1250) — compile-verified, matrix operand order checked by reading, not run on hardware
//
#include <hip/hip_runtime.h>
#include <hip/hip_bf16.h>

// ---------------------------------------------------------------------------
// Attention block on MI455X (gfx1250), wave32 + v_wmma_f32_16x16x32_f16.
//
// Pipeline (intermediates in d_ws; everything stays L2-resident):
//   1) qkv_pw_gemm : qkv_pw[768x4096]f32 = Wpw[768x256] @ x[256x4096] + b
//   2) dw_conv11   : 11x11 depthwise (pad 5) per channel -> qkv f16
//   3) flash_attn  : per head (d=32) online-softmax flash attention
//                    - transposed-score layout: S^T D-tile feeds the P^T
//                      B-operand IN-LANE (no cross-lane shuffle)
//                    - K tile cooperatively staged to LDS transposed
//                    - V tile staged by the Tensor Data Mover (async, double
//                      buffered, s_wait_tensorcnt before the barrier)
//   4) proj_gemm   : relu(Wlin @ attn + b) + x -> d_out (f32)
//
// WMMA register layouts (wave32, 16x16x32 f16), Lm=lane&15, hi=lane>>4:
//   A[16M x 32K]: row M=Lm; halves 0..7 -> K=hi*8+t, halves 8..15 -> K=16+hi*8+t
//   B[32K x 16N]: same map with N=Lm (mirror of A)
//   C/D[16M x 16N] f32: VGPR r holds M=r+hi*8, N=Lm
// ---------------------------------------------------------------------------

typedef __attribute__((ext_vector_type(16))) _Float16 v16h;
typedef __attribute__((ext_vector_type(8)))  _Float16 v8h;
typedef __attribute__((ext_vector_type(4)))  _Float16 h4;
typedef __attribute__((ext_vector_type(8)))  float    v8f;
typedef __attribute__((ext_vector_type(4)))  unsigned int v4u;
typedef __attribute__((ext_vector_type(8)))  int      v8i;
typedef __attribute__((ext_vector_type(4)))  int      v4i;

#define C_DIM  256
#define N_PIX  4096
#define QKV_C  768
#define HDIM   32
#define INV_TEMP 0.2f   // temp = float(int(sqrt(32))) = 5.0

#if __has_builtin(__builtin_amdgcn_tensor_load_to_lds) && \
    __has_builtin(__builtin_amdgcn_s_wait_tensorcnt)
#define HAVE_TDM 1
#else
#define HAVE_TDM 0
#endif

__device__ __forceinline__ v8f wmma_f16(v16h a, v16h b, v8f c) {
  return __builtin_amdgcn_wmma_f32_16x16x32_f16(
      /*neg_a=*/false, a, /*neg_b=*/false, b,
      /*c_mod=*/(short)0, c, /*reuse_a=*/false, /*reuse_b=*/false);
}

// ---------------------------------------------------------------------------
// Kernel 1: QKV pointwise conv: out[768][4096] = W[768][256] @ X[256][4096] + b
// ---------------------------------------------------------------------------
__global__ __launch_bounds__(256)
void qkv_pw_gemm(const float* __restrict__ X, const float* __restrict__ W,
                 const float* __restrict__ bias, float* __restrict__ out) {
  const int tid  = threadIdx.x;
  const int lane = tid & 31, wid = tid >> 5;
  const int tile = blockIdx.x * 8 + wid;      // 48 * 256 = 12288 tiles
  const int m0 = (tile >> 8) << 4;
  const int n0 = (tile & 255) << 4;
  const int Lm = lane & 15, hi = lane >> 4;

  v8f acc = {};
  const float* wrow = W + (m0 + Lm) * C_DIM;
  for (int kk = 0; kk < C_DIM; kk += 32) {
    v16h a, b;
    const float* wp = wrow + kk + hi * 8;     // two contiguous runs of 8
    #pragma unroll
    for (int t = 0; t < 8; ++t) {
      a[t]     = (_Float16)wp[t];
      a[8 + t] = (_Float16)wp[16 + t];
    }
    const float* xp = X + n0 + Lm;            // column n, strided over K
    #pragma unroll
    for (int t = 0; t < 8; ++t) {
      b[t]     = (_Float16)xp[(kk + hi * 8 + t) * N_PIX];
      b[8 + t] = (_Float16)xp[(kk + 16 + hi * 8 + t) * N_PIX];
    }
    acc = wmma_f16(a, b, acc);
  }
  #pragma unroll
  for (int r = 0; r < 8; ++r) {
    int row = m0 + hi * 8 + r;
    out[row * N_PIX + n0 + Lm] = acc[r] + bias[row];
  }
}

// ---------------------------------------------------------------------------
// Kernel 2: 11x11 depthwise conv, pad 5; one channel per block, LDS halo tile.
// ---------------------------------------------------------------------------
__global__ __launch_bounds__(256)
void dw_conv11(const float* __restrict__ in, const float* __restrict__ w,
               const float* __restrict__ bias, _Float16* __restrict__ out) {
  __shared__ float tile[74 * 74];
  __shared__ float wsh[121];
  const int c = blockIdx.x;
  const int tid = threadIdx.x;
  const float* src = in + c * N_PIX;

  for (int i = tid; i < 74 * 74; i += 256) {
    int r = i / 74, q = i - r * 74;
    int gr = r - 5, gc = q - 5;
    float v = 0.f;
    if (gr >= 0 && gr < 64 && gc >= 0 && gc < 64) v = src[gr * 64 + gc];
    tile[i] = v;
  }
  if (tid < 121) wsh[tid] = w[c * 121 + tid];
  __syncthreads();

  const float b = bias[c];
  #pragma unroll 1
  for (int i = 0; i < 16; ++i) {
    int p = tid + i * 256;
    int row = p >> 6, col = p & 63;
    float acc = b;
    #pragma unroll
    for (int ky = 0; ky < 11; ++ky) {
      const float* tr = &tile[(row + ky) * 74 + col];
      const float* wr = &wsh[ky * 11];
      #pragma unroll
      for (int kx = 0; kx < 11; ++kx) acc += tr[kx] * wr[kx];
    }
    out[c * N_PIX + p] = (_Float16)acc;
  }
}

// ---------------------------------------------------------------------------
// Kernel 3: flash attention (transposed-score).
//   S^T = (K A-tile)[16m x 32d] @ (Q B-tile)[32d x 16n]      1 WMMA, K-dim=d=32
//   OUT^T += (V A-tile)[16d' x 32m] @ (P^T B-tile)[32m x 16n]
// Block = 8 waves, one head, 128 queries; waves share K/V tiles via LDS.
//   kT[buf][m][d] : staged by all threads with in-flight transpose
//   vS[buf][d][m] : staged by the TDM (tensor_load_to_lds), wave 0 issues
// Double buffered, one barrier per 32-key iteration.
// ---------------------------------------------------------------------------
__global__ __launch_bounds__(256)
void flash_attn(const _Float16* __restrict__ qkv, _Float16* __restrict__ outh) {
  __shared__ __align__(16) _Float16 kT[2][32][32];   // [buf][m][d]
  __shared__ __align__(16) _Float16 vS[2][32][32];   // [buf][d][m]

  const int tid  = threadIdx.x;
  const int lane = tid & 31, wid = tid >> 5;
  const int h    = blockIdx.x >> 5;           // 8 heads
  const int nblk = blockIdx.x & 31;           // 32 query blocks of 128
  const int n0   = nblk * 128 + wid * 16;
  const int Lm = lane & 15, hi = lane >> 4;

  const _Float16* qh = qkv + (h * HDIM) * N_PIX;
  const _Float16* kh = qkv + (C_DIM + h * HDIM) * N_PIX;
  const _Float16* vh = qkv + (2 * C_DIM + h * HDIM) * N_PIX;

  const int sd = tid >> 3;          // 0..31: channel (d) row staged
  const int sm = (tid & 7) * 4;     // 0..28: key offset staged

  // loop-invariant Q B-operand: B[K=d][N=n] = q[d][n0+Lm]
  v16h bq;
  #pragma unroll
  for (int t = 0; t < 8; ++t) {
    bq[t]     = qh[(hi * 8 + t) * N_PIX + n0 + Lm];
    bq[8 + t] = qh[(16 + hi * 8 + t) * N_PIX + n0 + Lm];
  }

  // K tile staging with transpose: kT[buf][m][d] <- k[d][m0+m]
  auto stage_k = [&](int buf, int m0) {
    h4 kv = *(const h4*)(kh + sd * N_PIX + m0 + sm);
    #pragma unroll
    for (int j = 0; j < 4; ++j) kT[buf][sm + j][sd] = kv[j];
  };

#if HAVE_TDM
  // V tile via Tensor Data Mover: 2-D tensor [32 d][4096 m] (2B elements),
  // tile [dim1=32 rows][dim0=32 elems] -> LDS row-major vS[buf][d][m].
  auto issue_v = [&](int buf, int m0) {
    unsigned long long ga = (unsigned long long)(__SIZE_TYPE__)(vh + m0);
    unsigned int lds = (unsigned int)(__SIZE_TYPE__)(&vS[buf][0][0]);
    v4u g0 = { 1u,                                   // count=1, user descriptor
               lds,                                  // lds_addr
               (unsigned int)ga,                     // global_addr[31:0]
               ((unsigned int)(ga >> 32) & 0x1FFFFFFu) | 0x80000000u }; // +type=2
    v8i g1 = { (int)0x00010000,   // data_size=1 (2 bytes); wg_mask=0
               (int)0x10000000,   // tensor_dim0 = 4096 (lo16 << 16)
               (int)0x00200000,   // tensor_dim1 = 32   (lo16 << 16)
               (int)0x00200000,   // tile_dim0   = 32   (<< 16)
               (int)0x00000020,   // tile_dim1   = 32, tile_dim2 = 0
               (int)0x00001000,   // tensor_dim0_stride = 4096
               0, 0 };
    v4i z4 = {0, 0, 0, 0};
#if defined(__clang_major__) && __clang_major__ >= 23
    v8i z8 = {0, 0, 0, 0, 0, 0, 0, 0};
    __builtin_amdgcn_tensor_load_to_lds(g0, g1, z4, z4, z8, 0);
#else
    __builtin_amdgcn_tensor_load_to_lds(g0, g1, z4, z4, 0);
#endif
  };
#else
  auto stage_v = [&](int buf, int m0) {
    *(h4*)&vS[buf][sd][sm] = *(const h4*)(vh + sd * N_PIX + m0 + sm);
  };
#endif

  v8f acc0 = {}, acc1 = {};                   // OUT^T, d' 0-15 / 16-31
  float runM = -1e30f, runL = 0.f;

  // prologue: stage tile 0
  stage_k(0, 0);
#if HAVE_TDM
  if (wid == 0) {
    issue_v(0, 0);
    __builtin_amdgcn_s_wait_tensorcnt(0);
  }
#else
  stage_v(0, 0);
#endif
  __syncthreads();

  for (int it = 0; it < N_PIX / 32; ++it) {
    const int m0  = it * 32;
    const int buf = it & 1;

    // stage next tile into the back buffer (overlaps with compute below)
    if (it + 1 < N_PIX / 32) {
      stage_k(buf ^ 1, m0 + 32);
#if HAVE_TDM
      if (wid == 0) issue_v(buf ^ 1, m0 + 32);
#else
      stage_v(buf ^ 1, m0 + 32);
#endif
    }

    // ---- compute on current buffer ----
    const _Float16* kb = &kT[buf][0][0];
    const _Float16* vb = &vS[buf][0][0];

    // K^T A-operands from LDS: contiguous 16B runs (ds_load_b128)
    v8h k0a = *(const v8h*)(kb + Lm * 32 + hi * 8);
    v8h k0b = *(const v8h*)(kb + Lm * 32 + 16 + hi * 8);
    v8h k1a = *(const v8h*)(kb + (16 + Lm) * 32 + hi * 8);
    v8h k1b = *(const v8h*)(kb + (16 + Lm) * 32 + 16 + hi * 8);
    v16h ak0, ak1;
    #pragma unroll
    for (int t = 0; t < 8; ++t) {
      ak0[t] = k0a[t]; ak0[8 + t] = k0b[t];
      ak1[t] = k1a[t]; ak1[8 + t] = k1b[t];
    }
    v8f z = {};
    v8f s0 = wmma_f16(ak0, bq, z);            // m_local 0..15
    v8f s1 = wmma_f16(ak1, bq, z);            // m_local 16..31

    // online softmax over keys m (per column n = per lane)
    float tm = -1e30f;
    #pragma unroll
    for (int r = 0; r < 8; ++r) {
      s0[r] *= INV_TEMP; s1[r] *= INV_TEMP;
      tm = fmaxf(tm, fmaxf(s0[r], s1[r]));
    }
    tm = fmaxf(tm, __shfl_xor(tm, 16, 32));   // combine lane halves
    float newM = fmaxf(runM, tm);
    float corr = __expf(runM - newM);
    runM = newM;
    runL *= corr;
    #pragma unroll
    for (int r = 0; r < 8; ++r) { acc0[r] *= corr; acc1[r] *= corr; }

    // P^T B-operand, built in-lane from the S^T D-tiles
    v16h bp;
    #pragma unroll
    for (int r = 0; r < 8; ++r) {
      float p0 = __expf(s0[r] - newM);
      float p1 = __expf(s1[r] - newM);
      runL += p0 + p1;
      bp[r]     = (_Float16)p0;
      bp[8 + r] = (_Float16)p1;
    }

    // V A-operands from LDS (TDM-staged), contiguous 16B runs
    v8h va0 = *(const v8h*)(vb + Lm * 32 + hi * 8);
    v8h wb0 = *(const v8h*)(vb + Lm * 32 + 16 + hi * 8);
    v8h va1 = *(const v8h*)(vb + (16 + Lm) * 32 + hi * 8);
    v8h wb1 = *(const v8h*)(vb + (16 + Lm) * 32 + 16 + hi * 8);
    v16h av0, av1;
    #pragma unroll
    for (int t = 0; t < 8; ++t) {
      av0[t] = va0[t]; av0[8 + t] = wb0[t];
      av1[t] = va1[t]; av1[8 + t] = wb1[t];
    }
    acc0 = wmma_f16(av0, bp, acc0);
    acc1 = wmma_f16(av1, bp, acc1);

    // next tile must be fully in LDS before anyone enters iteration it+1
#if HAVE_TDM
    if (wid == 0) __builtin_amdgcn_s_wait_tensorcnt(0);
#endif
    __syncthreads();
  }

  float Ltot = runL + __shfl_xor(runL, 16, 32);
  float inv = 1.f / Ltot;
  #pragma unroll
  for (int r = 0; r < 8; ++r) {
    int d0 = hi * 8 + r;
    outh[(h * HDIM + d0) * N_PIX + n0 + Lm]      = (_Float16)(acc0[r] * inv);
    outh[(h * HDIM + 16 + d0) * N_PIX + n0 + Lm] = (_Float16)(acc1[r] * inv);
  }
}

// ---------------------------------------------------------------------------
// Kernel 4: proj GEMM + bias + ReLU + residual -> d_out (f32)
// ---------------------------------------------------------------------------
__global__ __launch_bounds__(256)
void proj_gemm(const _Float16* __restrict__ X, const float* __restrict__ W,
               const float* __restrict__ bias, const float* __restrict__ resid,
               float* __restrict__ out) {
  const int tid  = threadIdx.x;
  const int lane = tid & 31, wid = tid >> 5;
  const int tile = blockIdx.x * 8 + wid;      // 16 * 256 = 4096 tiles
  const int m0 = (tile >> 8) << 4;
  const int n0 = (tile & 255) << 4;
  const int Lm = lane & 15, hi = lane >> 4;

  v8f acc = {};
  const float* wrow = W + (m0 + Lm) * C_DIM;
  for (int kk = 0; kk < C_DIM; kk += 32) {
    v16h a, b;
    const float* wp = wrow + kk + hi * 8;
    #pragma unroll
    for (int t = 0; t < 8; ++t) {
      a[t]     = (_Float16)wp[t];
      a[8 + t] = (_Float16)wp[16 + t];
    }
    const _Float16* xp = X + n0 + Lm;
    #pragma unroll
    for (int t = 0; t < 8; ++t) {
      b[t]     = xp[(kk + hi * 8 + t) * N_PIX];
      b[8 + t] = xp[(kk + 16 + hi * 8 + t) * N_PIX];
    }
    acc = wmma_f16(a, b, acc);
  }
  #pragma unroll
  for (int r = 0; r < 8; ++r) {
    int row = m0 + hi * 8 + r;
    int idx = row * N_PIX + n0 + Lm;
    float v = acc[r] + bias[row];
    out[idx] = fmaxf(v, 0.f) + resid[idx];
  }
}

// ---------------------------------------------------------------------------
extern "C" void kernel_launch(void* const* d_in, const int* in_sizes, int n_in,
                              void* d_out, int out_size, void* d_ws, size_t ws_size,
                              hipStream_t stream) {
  const float* x        = (const float*)d_in[0];
  const float* w_qkv_pw = (const float*)d_in[1];
  const float* b_qkv_pw = (const float*)d_in[2];
  const float* w_qkv_dw = (const float*)d_in[3];
  const float* b_qkv_dw = (const float*)d_in[4];
  const float* w_lin    = (const float*)d_in[5];
  const float* b_lin    = (const float*)d_in[6];

  char* ws = (char*)d_ws;
  float*    qkv_pw = (float*)ws;                                   // 768*4096 f32 (12.6 MB)
  _Float16* qkv_h  = (_Float16*)(ws + (size_t)QKV_C * N_PIX * 4);  // 768*4096 f16 ( 6.3 MB)
  _Float16* attn_h = (_Float16*)(ws + (size_t)QKV_C * N_PIX * 6);  // 256*4096 f16 ( 2.1 MB)

  qkv_pw_gemm<<<1536, 256, 0, stream>>>(x, w_qkv_pw, b_qkv_pw, qkv_pw);
  dw_conv11  <<<QKV_C, 256, 0, stream>>>(qkv_pw, w_qkv_dw, b_qkv_dw, qkv_h);
  flash_attn <<<256,  256, 0, stream>>>(qkv_h, attn_h);
  proj_gemm  <<<512,  256, 0, stream>>>(attn_h, w_lin, b_lin, x, (float*)d_out);
}
